// NRIDecoder_15582141349983
// MI455X (gfx1250) — matrix operand order repaired and checked by reading, not compiled, not run on hardware
//
#include <hip/hip_runtime.h>

#define NN 10000
#define EE 160000
#define TT 32
#define HH 256
#define NF 192
#define TTEACH 24

#if __has_builtin(__builtin_amdgcn_global_load_async_to_lds_b128) && \
    __has_builtin(__builtin_amdgcn_s_wait_asynccnt)
#define USE_ASYNC_LDS 1
#else
#define USE_ASYNC_LDS 0
#endif

typedef __attribute__((ext_vector_type(16))) _Float16 v16h;
typedef __attribute__((ext_vector_type(8)))  _Float16 v8h;
typedef __attribute__((ext_vector_type(8)))  float    v8f;
typedef __attribute__((ext_vector_type(4)))  int      v4i;

typedef __attribute__((address_space(1))) v4i* gv4i_t;  // global <4 x i32>*
typedef __attribute__((address_space(3))) v4i* lv4i_t;  // LDS    <4 x i32>*

union HV { v16h v; v8h h8[2]; };

__device__ __forceinline__ v8f vz8() {
  v8f v = {0.f, 0.f, 0.f, 0.f, 0.f, 0.f, 0.f, 0.f};
  return v;
}

__device__ __forceinline__ v8f wmma32(v16h a, v16h b, v8f c) {
  return __builtin_amdgcn_wmma_f32_16x16x32_f16(false, a, false, b, (short)0, c,
                                                false, false);
}

// B fragment: packed per-lane 16 contiguous halves (two b128 loads)
__device__ __forceinline__ v16h load_b_frag(const _Float16* __restrict__ pack, int frag) {
  const int lane = threadIdx.x & 31;
  const _Float16* p = pack + ((size_t)frag * 32 + (size_t)lane) * 16;
  HV r;
  r.h8[0] = *(const v8h*)p;
  r.h8[1] = *(const v8h*)(p + 8);
  return r.v;
}

// A fragment from row-major f16 global: lane<16 row=lane K={0..7,16..23}; lane>=16 row=lane-16 K={8..15,24..31}
__device__ __forceinline__ v16h load_a_frag_g(const _Float16* __restrict__ base, int stride,
                                              int row0, int kk0) {
  const int lane = threadIdx.x & 31;
  const _Float16* p = base + (size_t)(row0 + (lane & 15)) * stride + kk0 + ((lane >> 4) << 3);
  HV r;
  r.h8[0] = *(const v8h*)p;
  r.h8[1] = *(const v8h*)(p + 16);
  return r.v;
}

// ---- pack f32 [Kdim,Ndim] (optionally transposed source) into WMMA B-fragment f16 layout
__global__ void pack_b_kernel(const float* __restrict__ src, _Float16* __restrict__ dst,
                              int Kdim, int Ndim, int transposed) {
  const int frag = blockIdx.x;
  const int lane = threadIdx.x;
  const int ntiles = Ndim >> 4;
  const int kk = frag / ntiles, nt = frag - kk * ntiles;
  const int col = nt * 16 + (lane & 15);
  const int kr0 = kk * 32 + ((lane >> 4) << 3);
  _Float16* o = dst + ((size_t)frag * 32 + lane) * 16;
  for (int h = 0; h < 16; ++h) {
    int kr = kr0 + (h < 8 ? h : h + 8);
    float v = transposed ? src[(size_t)col * Kdim + kr] : src[(size_t)kr * Ndim + col];
    o[h] = (_Float16)v;
  }
}

__global__ void zero_f32_kernel(float* __restrict__ p, int n) {
  int i = blockIdx.x * blockDim.x + threadIdx.x;
  if (i < n) p[i] = 0.f;
}

__global__ void count_kernel(const int* __restrict__ dstIdx, float* __restrict__ cnt) {
  int e = blockIdx.x * blockDim.x + threadIdx.x;
  if (e < EE) atomicAdd(&cnt[dstIdx[e]], 1.f);
}

__global__ void init_h_kernel(_Float16* __restrict__ h16) {
  int i = blockIdx.x * blockDim.x + threadIdx.x;
  if (i < NN * HH) h16[i] = (_Float16)0.f;
}

// fh16 = relu(h16 @ Wf + bf)   [NN,256]x[256,192] -> f16
__global__ __launch_bounds__(32) void fh_kernel(const _Float16* __restrict__ h16,
                                                const _Float16* __restrict__ Wfp,
                                                const float* __restrict__ bf,
                                                _Float16* __restrict__ fh16) {
  const int mt = blockIdx.x, nt = blockIdx.y, lane = threadIdx.x;
  v8f acc = vz8();
#pragma unroll
  for (int kk = 0; kk < 8; ++kk) {
    v16h a = load_a_frag_g(h16, HH, mt * 16, kk * 32);
    v16h b = load_b_frag(Wfp, kk * 12 + nt);
    acc = wmma32(a, b, acc);
  }
  const int cl = lane & 15;
  const int rb = mt * 16 + ((lane >> 4) << 3);
  const float bias = bf[nt * 16 + cl];
#pragma unroll
  for (int g = 0; g < 8; ++g) {
    float v = acc[g] + bias;
    v = v > 0.f ? v : 0.f;
    fh16[(size_t)(rb + g) * NF + nt * 16 + cl] = (_Float16)v;
  }
}

// gh = m16 @ Whh^T + bhh   [NN,256]x[256,768] -> f32
__global__ __launch_bounds__(32) void gh_kernel(const _Float16* __restrict__ m16,
                                                const _Float16* __restrict__ Whhp,
                                                const float* __restrict__ bhh,
                                                float* __restrict__ gh) {
  const int mt = blockIdx.x, nt = blockIdx.y, lane = threadIdx.x;
  v8f acc = vz8();
#pragma unroll
  for (int kk = 0; kk < 8; ++kk) {
    v16h a = load_a_frag_g(m16, HH, mt * 16, kk * 32);
    v16h b = load_b_frag(Whhp, kk * 48 + nt);
    acc = wmma32(a, b, acc);
  }
  const int cl = lane & 15;
  const int rb = mt * 16 + ((lane >> 4) << 3);
  const float bias = bhh[nt * 16 + cl];
#pragma unroll
  for (int g = 0; g < 8; ++g)
    gh[(size_t)(rb + g) * 768 + nt * 16 + cl] = acc[g] + bias;
}

// Fused edge MLP: one wave per 16-edge tile.
//   ef = [fh[dst], fh[src]]  staged via async global->LDS (ASYNCcnt) when available
//   for k in 0..2: h1 = relu(ef@W1k + b1k) * z_k  -> LDS
//                  msg += (h1 @ W2k) + z_k*b2k    (128 VGPR accumulators)
//   scatter-add msg rows to m[dst] with fp32 atomics
__global__ __launch_bounds__(32) void edge_msg_kernel(
    const _Float16* __restrict__ fh16, const int* __restrict__ ei,
    const float* __restrict__ z, const _Float16* __restrict__ W1p,
    const _Float16* __restrict__ W2p, const float* __restrict__ b1,
    const float* __restrict__ b2, float* __restrict__ m) {
  __shared__ __align__(16) _Float16 ef[16][392];   // 384 + 8 pad (bank-conflict-free)
  __shared__ __align__(16) _Float16 h1s[16][264];  // 256 + 8 pad
  __shared__ float zsh[3][16];

  const int lane = threadIdx.x;
  const int e0 = blockIdx.x * 16;
  const int* __restrict__ srcIdx = ei;
  const int* __restrict__ dstIdx = ei + EE;

  // stage ef tile: 16 edges x 48 b128-chunks (24 from fh[dst], 24 from fh[src])
  for (int i = lane; i < 16 * 48; i += 32) {
    const int e = i / 48;
    const int c = i - e * 48;
    const int node = (c < 24) ? dstIdx[e0 + e] : srcIdx[e0 + e];
    const int part = (c < 24) ? c : c - 24;
    const _Float16* gp = fh16 + (size_t)node * NF + part * 8;
#if USE_ASYNC_LDS
    _Float16* gpm = const_cast<_Float16*>(gp);
    __builtin_amdgcn_global_load_async_to_lds_b128(
        (gv4i_t)gpm, (lv4i_t)&ef[e][c * 8], /*offset=*/0, /*cpol=*/0);
#else
    *(v8h*)&ef[e][c * 8] = *(const v8h*)gp;
#endif
  }
  if (lane < 16) {
#pragma unroll
    for (int k = 0; k < 3; ++k) zsh[k][lane] = z[(size_t)(e0 + lane) * 4 + k + 1];
  }
#if USE_ASYNC_LDS
  __builtin_amdgcn_s_wait_asynccnt(0);
#endif
  __syncthreads();

  const int ebase = (lane >> 4) << 3;  // edge sub-block handled by this lane half
  const int cl = lane & 15;
  const int roff = (lane >> 4) << 3;   // A-fragment K sub-offset

  v8f msg[16];
#pragma unroll
  for (int nt = 0; nt < 16; ++nt) msg[nt] = vz8();

  for (int k = 0; k < 3; ++k) {
    const _Float16* W1k = W1p + (size_t)k * (12 * 16 * 512);
    const _Float16* W2k = W2p + (size_t)k * (8 * 16 * 512);
    float zl[8];
#pragma unroll
    for (int g = 0; g < 8; ++g) zl[g] = zsh[k][ebase + g];

    // GEMM1: [16,384] x [384,256]  ->  relu(.)*z  -> LDS
    for (int nt = 0; nt < 16; ++nt) {
      v8f acc = vz8();
#pragma unroll
      for (int kk = 0; kk < 12; ++kk) {
        HV a;
        a.h8[0] = *(const v8h*)&ef[cl][kk * 32 + roff];
        a.h8[1] = *(const v8h*)&ef[cl][kk * 32 + roff + 16];
        v16h b = load_b_frag(W1k, kk * 16 + nt);
        acc = wmma32(a.v, b, acc);
      }
      const float bias = b1[k * HH + nt * 16 + cl];
#pragma unroll
      for (int g = 0; g < 8; ++g) {
        float v = acc[g] + bias;
        v = v > 0.f ? v : 0.f;
        h1s[ebase + g][nt * 16 + cl] = (_Float16)(v * zl[g]);
      }
    }
    __syncthreads();

    // GEMM2: [16,256] x [256,256], accumulate across k into msg
#pragma unroll
    for (int nt = 0; nt < 16; ++nt) {
      const float b2v = b2[k * HH + nt * 16 + cl];
      v8f acc = msg[nt];
#pragma unroll
      for (int g = 0; g < 8; ++g) acc[g] += zl[g] * b2v;
#pragma unroll
      for (int kk = 0; kk < 8; ++kk) {
        HV a;
        a.h8[0] = *(const v8h*)&h1s[cl][kk * 32 + roff];
        a.h8[1] = *(const v8h*)&h1s[cl][kk * 32 + roff + 16];
        v16h b = load_b_frag(W2k, kk * 16 + nt);
        acc = wmma32(a.v, b, acc);
      }
      msg[nt] = acc;
    }
    __syncthreads();
  }

  int nd[8];
#pragma unroll
  for (int g = 0; g < 8; ++g) nd[g] = dstIdx[e0 + ebase + g];
#pragma unroll
  for (int nt = 0; nt < 16; ++nt) {
#pragma unroll
    for (int g = 0; g < 8; ++g)
      atomicAdd(&m[(size_t)nd[g] * HH + nt * 16 + cl], msg[nt][g]);
  }
}

// m /= max(cnt,1); also emit f16 copy for the gh GEMM
__global__ void mscale_kernel(float* __restrict__ m, _Float16* __restrict__ m16,
                              const float* __restrict__ cnt) {
  int i = blockIdx.x * blockDim.x + threadIdx.x;
  if (i >= NN * HH) return;
  float c = cnt[i >> 8];
  float v = m[i] / fmaxf(c, 1.f);
  m[i] = v;
  m16[i] = (_Float16)v;
}

// GRU gates + output head; one wave per node, 8 hidden channels per lane
__global__ __launch_bounds__(256) void node_kernel(
    const float* __restrict__ x, const float* __restrict__ gh,
    const float* __restrict__ m, const float* __restrict__ Wih,
    const float* __restrict__ bih, const float* __restrict__ Wo,
    const float* __restrict__ bo, _Float16* __restrict__ h16,
    float* __restrict__ out, int t) {
  const int wid = (blockIdx.x * blockDim.x + threadIdx.x) >> 5;
  const int lane = threadIdx.x & 31;
  if (wid >= NN) return;
  const int n = wid;

  float in6[6];
#pragma unroll
  for (int c = 0; c < 6; ++c)
    in6[c] = (t < TTEACH) ? x[(size_t)n * NF + t * 6 + c]
                          : out[(size_t)n * NF + (t - 1) * 6 + c];

  float part[6] = {0.f, 0.f, 0.f, 0.f, 0.f, 0.f};
#pragma unroll
  for (int i = 0; i < 8; ++i) {
    const int hh = lane * 8 + i;
    float gxr = bih[hh], gxz = bih[HH + hh], gxn = bih[2 * HH + hh];
#pragma unroll
    for (int c = 0; c < 6; ++c) {
      gxr += in6[c] * Wih[(size_t)hh * 6 + c];
      gxz += in6[c] * Wih[(size_t)(HH + hh) * 6 + c];
      gxn += in6[c] * Wih[(size_t)(2 * HH + hh) * 6 + c];
    }
    const float ghr = gh[(size_t)n * 768 + hh];
    const float ghz = gh[(size_t)n * 768 + HH + hh];
    const float ghn = gh[(size_t)n * 768 + 2 * HH + hh];
    const float r  = 1.f / (1.f + __expf(-(gxr + ghr)));
    const float zg = 1.f / (1.f + __expf(-(gxz + ghz)));
    const float nn = tanhf(gxn + r * ghn);
    const float mv = m[(size_t)n * HH + hh];
    const float hn = (1.f - zg) * nn + zg * mv;
    h16[(size_t)n * HH + hh] = (_Float16)hn;
#pragma unroll
    for (int c = 0; c < 6; ++c) part[c] += hn * Wo[(size_t)hh * 6 + c];
  }
#pragma unroll
  for (int off = 16; off > 0; off >>= 1) {
#pragma unroll
    for (int c = 0; c < 6; ++c) part[c] += __shfl_xor(part[c], off, 32);
  }
  if (lane == 0) {
#pragma unroll
    for (int c = 0; c < 6; ++c)
      out[(size_t)n * NF + t * 6 + c] = in6[c] + fmaxf(part[c] + bo[c], 0.f);
  }
}

extern "C" void kernel_launch(void* const* d_in, const int* in_sizes, int n_in,
                              void* d_out, int out_size, void* d_ws, size_t ws_size,
                              hipStream_t stream) {
  (void)in_sizes; (void)n_in; (void)out_size; (void)ws_size;
  const float* x   = (const float*)d_in[0];
  const int*   ei  = (const int*)d_in[1];
  const float* z   = (const float*)d_in[2];
  const float* Wf  = (const float*)d_in[3];
  const float* bf  = (const float*)d_in[4];
  const float* W1  = (const float*)d_in[5];
  const float* b1  = (const float*)d_in[6];
  const float* W2  = (const float*)d_in[7];
  const float* b2  = (const float*)d_in[8];
  const float* Wih = (const float*)d_in[9];
  const float* bih = (const float*)d_in[10];
  const float* Whh = (const float*)d_in[11];
  const float* bhh = (const float*)d_in[12];
  const float* Wo  = (const float*)d_in[13];
  const float* bo  = (const float*)d_in[14];
  float* out = (float*)d_out;

  char* w = (char*)d_ws;
  size_t off = 0;
  auto take = [&](size_t bytes) -> void* {
    void* p = w + off;
    off += (bytes + 255) & ~(size_t)255;
    return p;
  };
  _Float16* Wfp  = (_Float16*)take((size_t)8 * 12 * 512 * 2);
  _Float16* W1p  = (_Float16*)take((size_t)3 * 12 * 16 * 512 * 2);
  _Float16* W2p  = (_Float16*)take((size_t)3 * 8 * 16 * 512 * 2);
  _Float16* Whhp = (_Float16*)take((size_t)8 * 48 * 512 * 2);
  _Float16* h16  = (_Float16*)take((size_t)NN * HH * 2);
  _Float16* fh16 = (_Float16*)take((size_t)NN * NF * 2);
  float*    mbuf = (float*)   take((size_t)NN * HH * 4);
  _Float16* m16  = (_Float16*)take((size_t)NN * HH * 2);
  float*    ghb  = (float*)   take((size_t)NN * 768 * 4);
  float*    cnt  = (float*)   take((size_t)NN * 4);

  // weight prepack (f16, WMMA B-fragment layout)
  pack_b_kernel<<<96, 32, 0, stream>>>(Wf, Wfp, HH, NF, 0);
  for (int k = 0; k < 3; ++k) {
    pack_b_kernel<<<192, 32, 0, stream>>>(W1 + (size_t)k * 384 * HH,
                                          W1p + (size_t)k * 12 * 16 * 512, 384, HH, 0);
    pack_b_kernel<<<128, 32, 0, stream>>>(W2 + (size_t)k * HH * HH,
                                          W2p + (size_t)k * 8 * 16 * 512, HH, HH, 0);
  }
  pack_b_kernel<<<384, 32, 0, stream>>>(Whh, Whhp, HH, 768, 1);

  zero_f32_kernel<<<(NN + 255) / 256, 256, 0, stream>>>(cnt, NN);
  count_kernel<<<(EE + 255) / 256, 256, 0, stream>>>(ei + EE, cnt);
  init_h_kernel<<<(NN * HH + 255) / 256, 256, 0, stream>>>(h16);

  for (int t = 0; t < TT; ++t) {
    fh_kernel<<<dim3(NN / 16, NF / 16), 32, 0, stream>>>(h16, Wfp, bf, fh16);
    zero_f32_kernel<<<(NN * HH + 255) / 256, 256, 0, stream>>>(mbuf, NN * HH);
    edge_msg_kernel<<<EE / 16, 32, 0, stream>>>(fh16, ei, z, W1p, W2p, b1, b2, mbuf);
    mscale_kernel<<<(NN * HH + 255) / 256, 256, 0, stream>>>(mbuf, m16, cnt);
    gh_kernel<<<dim3(NN / 16, 768 / 16), 32, 0, stream>>>(m16, Whhp, bhh, ghb);
    node_kernel<<<(NN * 32 + 255) / 256, 256, 0, stream>>>(x, ghb, mbuf, Wih, bih,
                                                           Wo, bo, h16, out, t);
  }
}
